// MultiHeadAttention_53257594470450
// MI455X (gfx1250) — matrix operand order repaired and checked by reading, not compiled, not run on hardware
//
#include <hip/hip_runtime.h>
#include <hip/hip_bf16.h>

typedef __attribute__((ext_vector_type(16))) _Float16 v16h;
typedef __attribute__((ext_vector_type(8)))  _Float16 v8h;
typedef __attribute__((ext_vector_type(8)))  float    v8f;

#define S_DIM 4096
#define E_DIM 768
#define H_NUM 12
#define D_DIM 64

// LDS tile strides (halves), padded so 16 row-fanned lanes hit distinct banks
#define LDK_STRIDE 72   // K tile: 32 rows x 64 d
#define LDV_STRIDE 40   // V^T tile: 64 rows x 32 keys

// Async global->LDS staging (ASYNCcnt path), if this toolchain exposes it.
#if defined(__has_builtin)
#if __has_builtin(__builtin_amdgcn_global_load_async_to_lds_b128) && \
    __has_builtin(__builtin_amdgcn_s_wait_asynccnt)
#define USE_ASYNC_LDS 1
#endif
#endif
#ifndef USE_ASYNC_LDS
#define USE_ASYNC_LDS 0
#endif

#if USE_ASYNC_LDS
// Builtin signature (from hipcc diagnostic): params are pointers to
// int __attribute__((vector_size(4*sizeof(int)))), global src / LDS dst.
typedef int v4i_t __attribute__((vector_size(4 * sizeof(int))));
typedef __attribute__((address_space(1))) v4i_t g_v4i;
typedef __attribute__((address_space(3))) v4i_t l_v4i;
#endif

// ---------------------------------------------------------------------------
// Fragment helpers for V_WMMA_F32_16X16X32_F16 (wave32).
//
// A operand (16 M x 32 K), per-lane 16 halves:
//   lanes 0-15 : row m=lane,     halves = K 0..15 (contiguous)
//   lanes 16-31: row m=lane-16,  halves = K 16..31 (contiguous)
// B operand (32 K x 16 N), per-lane 16 halves:
//   lanes 0-15 : col n=lane,     halves = K 0..15 (contiguous)
//   lanes 16-31: col n=lane-16,  halves = K 16..31 (contiguous)
// C/D (16 M x 16 N) f32, 8 VGPRs:
//   lanes 0-15 : n=lane,    vgpr r -> m = r
//   lanes 16-31: n=lane-16, vgpr r -> m = r + 8
// ---------------------------------------------------------------------------

__device__ __forceinline__ v16h make16(v8h lo, v8h hi) {
  v16h r;
#pragma unroll
  for (int i = 0; i < 8; i++) { r[i] = lo[i]; r[i + 8] = hi[i]; }
  return r;
}

// p points at (row, k0) in a row-major [*, K] array, K-contiguous.
__device__ __forceinline__ v16h loadA(const _Float16* p, bool hiGrp) {
  const int o = hiGrp ? 16 : 0;
  v8h a = *(const v8h*)(p + o);
  v8h b = *(const v8h*)(p + o + 8);
  return make16(a, b);
}

// p points at (col, k0) in a col-major-as-[N][K] array, K-contiguous.
__device__ __forceinline__ v16h loadB(const _Float16* p, bool hiGrp) {
  const int o = hiGrp ? 16 : 0;
  v8h a = *(const v8h*)(p + o);
  v8h b = *(const v8h*)(p + o + 8);
  return make16(a, b);
}

__device__ __forceinline__ v8f wmma_f16(v16h a, v16h b, v8f c) {
  return __builtin_amdgcn_wmma_f32_16x16x32_f16(false, a, false, b, (short)0, c,
                                                false, false);
}

// ---------------------------------------------------------------------------
// Stage 0: f32 -> f16 conversion / weight transpose
// ---------------------------------------------------------------------------
__global__ void cvt_f32_to_f16(const float* __restrict__ in,
                               _Float16* __restrict__ out, int n) {
  int i = blockIdx.x * 256 + threadIdx.x;
  if (i < n) out[i] = (_Float16)in[i];
}

// W is [E][E] row-major (applied as x @ W); WT[n][k] = W[k][n]
__global__ void cvt_transpose_f16(const float* __restrict__ W,
                                  _Float16* __restrict__ WT) {
  int idx = blockIdx.x * 256 + threadIdx.x;
  if (idx >= E_DIM * E_DIM) return;
  int n = idx / E_DIM;
  int k = idx - n * E_DIM;
  WT[idx] = (_Float16)W[(size_t)k * E_DIM + n];  // coalesced write
}

// ---------------------------------------------------------------------------
// Stage 1/3: GEMM  C[M,N] = A[M,K] (f16) @ BT[N,K]^T (f16)
//   mode 0: outH[M][N] f16 row-major           (Q, K projections)
//   mode 1: outH[N][M] f16 transposed          (V projection -> V^T [E][S])
//   mode 2: outF[M][N] f32 + bias[N]           (final output projection)
// Block: 256 thr = 8 waves, block tile 256(M) x 64(N); wave tile 64 x 32.
// ---------------------------------------------------------------------------
__global__ __launch_bounds__(256) void gemm_wmma(
    const _Float16* __restrict__ A, const _Float16* __restrict__ BT,
    const float* __restrict__ bias, _Float16* __restrict__ outH,
    float* __restrict__ outF, int M, int N, int K, int mode) {
  const int lane = threadIdx.x & 31;
  const bool hi = lane >= 16;
  const int w = threadIdx.x >> 5;
  const int mBase = blockIdx.x * 256 + (w & 3) * 64;
  const int nBase = blockIdx.y * 64 + (w >> 2) * 32;

  v8f acc[4][2];
#pragma unroll
  for (int mt = 0; mt < 4; mt++)
#pragma unroll
    for (int nt = 0; nt < 2; nt++)
#pragma unroll
      for (int r = 0; r < 8; r++) acc[mt][nt][r] = 0.f;

  const int aRow0 = mBase + (lane & 15);
  const int bCol0 = nBase + (lane & 15);

  for (int kk = 0; kk < K; kk += 32) {
    v16h af[4], bf[2];
#pragma unroll
    for (int mt = 0; mt < 4; mt++)
      af[mt] = loadA(A + (size_t)(aRow0 + mt * 16) * K + kk, hi);
#pragma unroll
    for (int nt = 0; nt < 2; nt++)
      bf[nt] = loadB(BT + (size_t)(bCol0 + nt * 16) * K + kk, hi);
#pragma unroll
    for (int mt = 0; mt < 4; mt++)
#pragma unroll
      for (int nt = 0; nt < 2; nt++)
        acc[mt][nt] = wmma_f16(af[mt], bf[nt], acc[mt][nt]);
  }

  const int nCol = (lane & 15);
#pragma unroll
  for (int mt = 0; mt < 4; mt++) {
#pragma unroll
    for (int nt = 0; nt < 2; nt++) {
      const int col = nBase + nt * 16 + nCol;
      const int mTile = mBase + mt * 16;
      if (mode == 0) {
#pragma unroll
        for (int r = 0; r < 8; r++) {
          int m = mTile + r + (hi ? 8 : 0);
          outH[(size_t)m * N + col] = (_Float16)acc[mt][nt][r];
        }
      } else if (mode == 1) {
        // transposed store: out[col][m], 8 consecutive m per lane -> 16B store
        v8h ph;
#pragma unroll
        for (int r = 0; r < 8; r++) ph[r] = (_Float16)acc[mt][nt][r];
        *(v8h*)(outH + (size_t)col * M + mTile + (hi ? 8 : 0)) = ph;
      } else {
        float bv = bias[col];
#pragma unroll
        for (int r = 0; r < 8; r++) {
          int m = mTile + r + (hi ? 8 : 0);
          outF[(size_t)m * N + col] = acc[mt][nt][r] + bv;
        }
      }
    }
  }
}

// ---------------------------------------------------------------------------
// Stage 2: causal flash attention with block-cooperative K/V LDS staging.
// Block = 8 waves = 256 consecutive queries of ONE head. Each 32-key K-tile
// (32x64 f16) and V^T-tile (64x32 f16) is staged into LDS once per block and
// consumed by all 8 waves (8x L2 traffic reduction vs per-wave streaming).
// Staging uses GLOBAL_LOAD_ASYNC_TO_LDS_B128 (ASYNCcnt) when available.
// All waves iterate the block-uniform key range (barrier-safe) and skip the
// compute phase past their causal limit.
// Math: S^T = K.Q^T and O^T = V^T.P^T so softmax reductions are in-lane
// (+ one shfl_xor(16)); P^T repack to f16 B-operand needs only shfl_xor(16).
// ---------------------------------------------------------------------------
__global__ __launch_bounds__(256) void flash_attn(
    const _Float16* __restrict__ Q, const _Float16* __restrict__ Km,
    const _Float16* __restrict__ VT, _Float16* __restrict__ ctx) {
  const int lane = threadIdx.x & 31;
  const bool hi = lane >= 16;
  const int w = threadIdx.x >> 5;
  const int nQB = S_DIM / 256;              // 16 query-blocks per head
  const int head = blockIdx.x / nQB;
  const int qB = (blockIdx.x % nQB) * 256;  // block's first query
  const int qb = qB + w * 32;               // wave's first query
  const int eh = head * D_DIM;

  __shared__ _Float16 ldsK[32 * LDK_STRIDE];      // [key][d], padded
  __shared__ _Float16 ldsV[D_DIM * LDV_STRIDE];   // [d][key], padded

  // Q as B-operand fragments [qtile][d-chunk] (private, read once)
  v16h bq[2][2];
  const int qRow = qb + (lane & 15);
#pragma unroll
  for (int qt = 0; qt < 2; qt++)
#pragma unroll
    for (int dc = 0; dc < 2; dc++)
      bq[qt][dc] =
          loadB(Q + (size_t)(qRow + qt * 16) * E_DIM + eh + dc * 32, hi);

  v8f acc[2][4];  // O^T tiles [qtile][d-tile]
#pragma unroll
  for (int qt = 0; qt < 2; qt++)
#pragma unroll
    for (int dt = 0; dt < 4; dt++)
#pragma unroll
      for (int r = 0; r < 8; r++) acc[qt][dt][r] = 0.f;

  float mrun[2] = {-1e30f, -1e30f};
  float lrun[2] = {0.f, 0.f};

  // staging assignments (block-uniform)
  const int t = threadIdx.x;
  const int kR = t >> 3, kC = (t & 7) * 8;  // K tile: 32 rows x 64 halves
  const int vR = t >> 2, vC = (t & 3) * 8;  // V^T tile: 64 rows x 32 halves

  const int kbEnd = qB + 224;  // causal limit of the block's last wave
  for (int kb = 0; kb <= kbEnd; kb += 32) {
    // cooperative stage: 16B per thread per tile
#if USE_ASYNC_LDS
    {
      __builtin_amdgcn_global_load_async_to_lds_b128(
          (g_v4i*)(Km + (size_t)(kb + kR) * E_DIM + eh + kC),
          (l_v4i*)(ldsK + kR * LDK_STRIDE + kC), 0, 0);
      __builtin_amdgcn_global_load_async_to_lds_b128(
          (g_v4i*)(VT + (size_t)(eh + vR) * S_DIM + kb + vC),
          (l_v4i*)(ldsV + vR * LDV_STRIDE + vC), 0, 0);
      __builtin_amdgcn_s_wait_asynccnt(0);
    }
#else
    *(v8h*)(ldsK + kR * LDK_STRIDE + kC) =
        *(const v8h*)(Km + (size_t)(kb + kR) * E_DIM + eh + kC);
    *(v8h*)(ldsV + vR * LDV_STRIDE + vC) =
        *(const v8h*)(VT + (size_t)(eh + vR) * S_DIM + kb + vC);
#endif
    __syncthreads();

    if (kb <= qb) {
      // K as A-operand fragments [ktile][d-chunk] from LDS
      v16h ak[2][2];
#pragma unroll
      for (int kt = 0; kt < 2; kt++)
#pragma unroll
        for (int dc = 0; dc < 2; dc++)
          ak[kt][dc] =
              loadA(ldsK + (kt * 16 + (lane & 15)) * LDK_STRIDE + dc * 32, hi);

      // V^T as A-operand fragments [d-tile] from LDS
      v16h av[4];
#pragma unroll
      for (int dt = 0; dt < 4; dt++)
        av[dt] = loadA(ldsV + (dt * 16 + (lane & 15)) * LDV_STRIDE, hi);

      // S^T tiles: M=key, N=query, contraction over d (2 x 32)
      v8f st[2][2];
#pragma unroll
      for (int kt = 0; kt < 2; kt++)
#pragma unroll
        for (int qt = 0; qt < 2; qt++) {
          v8f z;
#pragma unroll
          for (int r = 0; r < 8; r++) z[r] = 0.f;
          z = wmma_f16(ak[kt][0], bq[qt][0], z);
          z = wmma_f16(ak[kt][1], bq[qt][1], z);
          st[kt][qt] = z;
        }

      // scale + causal mask (only the diagonal block can have key > query)
      const bool diag = (kb == qb);
#pragma unroll
      for (int kt = 0; kt < 2; kt++)
#pragma unroll
        for (int qt = 0; qt < 2; qt++)
#pragma unroll
          for (int r = 0; r < 8; r++) {
            float s = st[kt][qt][r] * 0.125f;  // 1/sqrt(64)
            if (diag) {
              int key = kb + kt * 16 + r + (hi ? 8 : 0);
              int q = qb + qt * 16 + (lane & 15);
              if (key > q) s = -1e30f;
            }
            st[kt][qt][r] = s;
          }

#pragma unroll
      for (int qt = 0; qt < 2; qt++) {
        // per-query (per-lane) running max
        float sm = -1e30f;
#pragma unroll
        for (int kt = 0; kt < 2; kt++)
#pragma unroll
          for (int r = 0; r < 8; r++) sm = fmaxf(sm, st[kt][qt][r]);
        sm = fmaxf(sm, __shfl_xor(sm, 16, 32));
        float mnew = fmaxf(mrun[qt], sm);
        float corr = __expf(mrun[qt] - mnew);
        mrun[qt] = mnew;

        float rsum = 0.f;
#pragma unroll
        for (int kt = 0; kt < 2; kt++)
#pragma unroll
          for (int r = 0; r < 8; r++) {
            float p = __expf(st[kt][qt][r] - mnew);
            st[kt][qt][r] = p;
            rsum += p;
          }
        rsum += __shfl_xor(rsum, 16, 32);
        lrun[qt] = lrun[qt] * corr + rsum;

#pragma unroll
        for (int dt = 0; dt < 4; dt++)
#pragma unroll
          for (int r = 0; r < 8; r++) acc[qt][dt][r] *= corr;

        // Repack P^T (f32 C-layout) into f16 B-operand layout:
        // lo lanes need keys 0..15, hi lanes keys 16..31 (relative to kb).
        v16h pf;
#pragma unroll
        for (int r = 0; r < 8; r++) {
          float o0 = __shfl_xor(st[0][qt][r], 16, 32);
          float o1 = __shfl_xor(st[1][qt][r], 16, 32);
          float e0 = hi ? o1 : st[0][qt][r];  // halves[r]
          float e1 = hi ? st[1][qt][r] : o0;  // halves[r+8]
          pf[r] = (_Float16)e0;
          pf[r + 8] = (_Float16)e1;
        }

#pragma unroll
        for (int dt = 0; dt < 4; dt++)
          acc[qt][dt] = wmma_f16(av[dt], pf, acc[qt][dt]);
      }
    }
    __syncthreads();  // all waves done reading tiles before next stage
  }

  // normalize and store ctx[s][e] (e = head*64 + d), 16B packed stores
#pragma unroll
  for (int qt = 0; qt < 2; qt++) {
    float inv = 1.f / lrun[qt];
    int srow = qb + qt * 16 + (lane & 15);
#pragma unroll
    for (int dt = 0; dt < 4; dt++) {
      v8h ph;
#pragma unroll
      for (int r = 0; r < 8; r++) ph[r] = (_Float16)(acc[qt][dt][r] * inv);
      *(v8h*)(ctx + (size_t)srow * E_DIM + eh + dt * 16 + (hi ? 8 : 0)) = ph;
    }
  }
}

// ---------------------------------------------------------------------------
// Host-side launcher
// ---------------------------------------------------------------------------
extern "C" void kernel_launch(void* const* d_in, const int* in_sizes, int n_in,
                              void* d_out, int out_size, void* d_ws,
                              size_t ws_size, hipStream_t stream) {
  const float* x = (const float*)d_in[0];
  const float* Wq = (const float*)d_in[1];
  const float* Wk = (const float*)d_in[2];
  const float* Wv = (const float*)d_in[3];
  const float* Wo = (const float*)d_in[4];
  const float* bo = (const float*)d_in[5];
  // d_in[6] = mask: causal handled analytically.

  const size_t SE = (size_t)S_DIM * E_DIM;
  const size_t EE = (size_t)E_DIM * E_DIM;

  _Float16* ws = (_Float16*)d_ws;
  _Float16* xh = ws;             // [S][E]
  _Float16* WqT = xh + SE;       // [E][E] transposed
  _Float16* WkT = WqT + EE;
  _Float16* WvT = WkT + EE;
  _Float16* WoT = WvT + EE;
  _Float16* Qh = WoT + EE;       // [S][E]
  _Float16* Kh = Qh + SE;        // [S][E]
  _Float16* VTh = Kh + SE;       // [E][S]
  _Float16* ctx = VTh + SE;      // [S][E]

  // Stage 0: conversions
  cvt_f32_to_f16<<<(int)((SE + 255) / 256), 256, 0, stream>>>(x, xh, (int)SE);
  int wgrid = (int)((EE + 255) / 256);
  cvt_transpose_f16<<<wgrid, 256, 0, stream>>>(Wq, WqT);
  cvt_transpose_f16<<<wgrid, 256, 0, stream>>>(Wk, WkT);
  cvt_transpose_f16<<<wgrid, 256, 0, stream>>>(Wv, WvT);
  cvt_transpose_f16<<<wgrid, 256, 0, stream>>>(Wo, WoT);

  // Stage 1: Q/K/V projections (V stored transposed)
  dim3 gg(S_DIM / 256, E_DIM / 64);
  gemm_wmma<<<gg, 256, 0, stream>>>(xh, WqT, nullptr, Qh, nullptr, S_DIM,
                                    E_DIM, E_DIM, 0);
  gemm_wmma<<<gg, 256, 0, stream>>>(xh, WkT, nullptr, Kh, nullptr, S_DIM,
                                    E_DIM, E_DIM, 0);
  gemm_wmma<<<gg, 256, 0, stream>>>(xh, WvT, nullptr, VTh, nullptr, S_DIM,
                                    E_DIM, E_DIM, 1);

  // Stage 2: causal flash attention (16 q-blocks x 12 heads = 192 blocks)
  flash_attn<<<(S_DIM / 256) * H_NUM, 256, 0, stream>>>(Qh, Kh, VTh, ctx);

  // Stage 3: output projection, f32 + bias
  gemm_wmma<<<gg, 256, 0, stream>>>(ctx, WoT, bo, nullptr, (float*)d_out,
                                    S_DIM, E_DIM, E_DIM, 2);
}
